// CrossAttn_5970004541838
// MI455X (gfx1250) — matrix operand order repaired and checked by reading, not compiled, not run on hardware
//
#include <hip/hip_runtime.h>
#include <cstdint>

typedef __attribute__((ext_vector_type(16))) __bf16 v16bf;
typedef __attribute__((ext_vector_type(8)))  __bf16 bf16x8;
typedef __attribute__((ext_vector_type(8)))  float  v8f;
typedef __attribute__((ext_vector_type(16))) unsigned char uchar16;

#define NUM_HEAD 16
#define ADIM     64
#define HSIZE    1024
#define ISIZE    1024
#define OSIZE    1024
#define BATCH    4
#define SEQ      1024
#define NQ       1024
#define NEG_INF  (-1000000000.0f)
#define EPSF     1e-9f

#define LDS_STRIDE 40   // bf16 elements per LDS row (32 data + 8 pad); 80B row stride

// Assemble a v16bf fragment from two 16-byte chunks.
static __device__ __forceinline__ v16bf make_frag(const __bf16* p0, const __bf16* p1) {
  bf16x8 lo = *(const bf16x8*)p0;
  bf16x8 hi = *(const bf16x8*)p1;
  v16bf r;
#pragma unroll
  for (int e = 0; e < 8; ++e) { r[e] = lo[e]; r[e + 8] = hi[e]; }
  return r;
}

// ---------------------------------------------------------------------------
// fp32 -> bf16 conversion (grid-stride)
// ---------------------------------------------------------------------------
__global__ void cvt_bf16(const float* __restrict__ in, __bf16* __restrict__ out, int n) {
  int i = blockIdx.x * blockDim.x + threadIdx.x;
  int stride = gridDim.x * blockDim.x;
  for (; i < n; i += stride) out[i] = (__bf16)in[i];
}

// ---------------------------------------------------------------------------
// C[M,N] = A[M,K] @ B[N,K]^T   (A,B bf16 row-major; C f32 or bf16)
// Block tile 128x128, 8 waves in 4(m) x 2(n); wave tile 32x64 = 2x4 WMMA tiles.
// Double-buffered LDS staging: next K-slab global loads overlap current WMMAs.
// ---------------------------------------------------------------------------
#define BM 128
#define BN 128
#define BK 32

template <bool F32OUT>
__global__ __launch_bounds__(256) void gemm_nt(
    const __bf16* __restrict__ A, const __bf16* __restrict__ B,
    float* __restrict__ Cf, __bf16* __restrict__ Cb,
    int M, int N, int K) {
  __shared__ alignas(16) __bf16 As[2][BM * LDS_STRIDE];
  __shared__ alignas(16) __bf16 Bs[2][BN * LDS_STRIDE];

  const int tid  = threadIdx.x;
  const int lane = tid & 31;
  const int wid  = tid >> 5;
  const int wm   = wid >> 1;            // 0..3  (32-row slice)
  const int wn   = wid & 1;             // 0..1  (64-col slice)
  const size_t m0 = (size_t)blockIdx.x * BM;
  const size_t n0 = (size_t)blockIdx.y * BN;
  const int lr  = lane & 15;
  const int kbA = (lane >> 4) << 3;     // A-frag: interleaved 8-wide K sub-block
  const int kbB = (lane >> 4) << 4;     // unused; B-frag also uses kbA (interleaved)

  v8f acc[2][4] = {};

  // Stage K-slab k0 into buffer buf: A/B each 128x32 bf16 = 512 16B-chunks,
  // 2 chunks per thread per matrix.
  auto stage = [&](int k0, int buf) {
#pragma unroll
    for (int c = 0; c < 2; ++c) {
      int ch = tid + c * 256;           // 0..511
      int row = ch >> 2, kc = (ch & 3) << 3;
      *(bf16x8*)&As[buf][row * LDS_STRIDE + kc] =
          *(const bf16x8*)&A[(m0 + row) * K + k0 + kc];
      *(bf16x8*)&Bs[buf][row * LDS_STRIDE + kc] =
          *(const bf16x8*)&B[(n0 + row) * K + k0 + kc];
    }
  };

  stage(0, 0);
  int buf = 0;
  for (int k0 = 0; k0 < K; k0 += BK, buf ^= 1) {
    __syncthreads();                    // staging into `buf` complete
    if (k0 + BK < K) stage(k0 + BK, buf ^ 1);   // overlap with WMMAs below
    if (k0 + 2 * BK < K) {
      __builtin_prefetch(&A[(m0 + (tid >> 1)) * K + k0 + 2 * BK], 0, 1);
      __builtin_prefetch(&B[(n0 + (tid >> 1)) * K + k0 + 2 * BK], 0, 1);
    }

    v16bf af[2], bfr[4];
#pragma unroll
    for (int i = 0; i < 2; ++i) {
      int row = wm * 32 + i * 16 + lr;
      af[i] = make_frag(&As[buf][row * LDS_STRIDE + kbA],
                        &As[buf][row * LDS_STRIDE + 16 + kbA]);
    }
#pragma unroll
    for (int j = 0; j < 4; ++j) {
      int row = wn * 64 + j * 16 + lr;
      bfr[j] = make_frag(&Bs[buf][row * LDS_STRIDE + kbA],
                         &Bs[buf][row * LDS_STRIDE + 16 + kbA]);
    }
#pragma unroll
    for (int i = 0; i < 2; ++i)
#pragma unroll
      for (int j = 0; j < 4; ++j)
        acc[i][j] = __builtin_amdgcn_wmma_f32_16x16x32_bf16(
            false, af[i], false, bfr[j], (short)0, acc[i][j], false, false);
  }

  const int mh = (lane >> 4) << 3;      // D-frag row offset for hi half-wave
#pragma unroll
  for (int i = 0; i < 2; ++i)
#pragma unroll
    for (int j = 0; j < 4; ++j)
#pragma unroll
      for (int r = 0; r < 8; ++r) {
        size_t m = m0 + wm * 32 + i * 16 + r + mh;
        size_t n = n0 + wn * 64 + j * 16 + lr;
        if (F32OUT) Cf[m * N + n] = acc[i][j][r];
        else        Cb[m * N + n] = (__bf16)acc[i][j][r];
      }
}

// ---------------------------------------------------------------------------
// Fused squared-ReLU attention.
//   grid = (BATCH*NUM_HEAD, NQ/64); block = 128 (4 waves, 16 q-rows each)
//   o_unnorm[q,:] = sum_s t(q,s) * V[s,:];  denom[q] = sum_s t(q,s)
//   t = relu(QK^T/8 (+mask) + nbias)^2 ;  o = o_unnorm / (denom + eps)
// ---------------------------------------------------------------------------
__global__ __launch_bounds__(128) void attn_sqrelu(
    const __bf16* __restrict__ qb,          // (B*NQ, HSIZE)
    const __bf16* __restrict__ kvb,         // (B*SEQ, 2*HSIZE): [k | v]
    const unsigned char* __restrict__ mask, // (B, NQ, SEQ) bool
    const float* __restrict__ nbias,
    __bf16* __restrict__ ob) {              // (B*NQ, HSIZE)
  __shared__ alignas(16) __bf16 Vt[ADIM * LDS_STRIDE];   // V chunk transposed [d][s]
  __shared__ alignas(16) __bf16 Ts[4][16 * LDS_STRIDE];  // per-wave t tile (16x32)
  __shared__ alignas(16) unsigned char Ms[64 * 32];      // mask tile (64 q x 32 s)

  const int tid  = threadIdx.x;
  const int lane = tid & 31;
  const int wid  = tid >> 5;
  const int h    = blockIdx.x & (NUM_HEAD - 1);
  const int bb   = blockIdx.x >> 4;
  const int qblk = blockIdx.y * 64;
  const int q0   = qblk + wid * 16;
  const int lr   = lane & 15;
  const int kbA  = (lane >> 4) << 3;
  const int kbB  = (lane >> 4) << 4;
  const int mh   = (lane >> 4) << 3;
  const float nb = nbias[0];
  const float scale = 0.125f;                  // 1/sqrt(ADIM)

  // Resident Q A-frags: contraction dim d = 0..31 (p=0), 32..63 (p=1)
  v16bf qa[2];
  {
    const __bf16* qrow = qb + ((size_t)(bb * NQ + q0 + lr)) * HSIZE + h * ADIM;
#pragma unroll
    for (int p = 0; p < 2; ++p)
      qa[p] = make_frag(qrow + p * 32 + kbA, qrow + p * 32 + 16 + kbA);
  }

  v8f oacc[4] = {};
  float dsum[8] = {};
  const unsigned char* mbase = mask + (size_t)bb * NQ * SEQ;

  for (int s0 = 0; s0 < SEQ; s0 += 32) {
    __syncthreads();
    // Stage V chunk transposed into LDS: 32 s-rows x 64 dims
#pragma unroll
    for (int i = 0; i < 2; ++i) {
      int c = tid + i * 128;                   // 256 chunks of 8 elements
      int srow = c >> 3, dc = (c & 7) << 3;
      bf16x8 vv = *(const bf16x8*)&kvb[((size_t)(bb * SEQ + s0 + srow)) * (2 * HSIZE)
                                       + HSIZE + h * ADIM + dc];
#pragma unroll
      for (int e = 0; e < 8; ++e) Vt[(dc + e) * LDS_STRIDE + srow] = vv[e];
    }
    // Stage mask tile: 64 q-rows x 32 s-cols, one 16B load per thread
    {
      int row = tid >> 1, c16 = (tid & 1) << 4;
      *(uchar16*)&Ms[row * 32 + c16] =
          *(const uchar16*)&mbase[(size_t)(qblk + row) * SEQ + s0 + c16];
    }
    __syncthreads();

    // Two 16-column score tiles per chunk
#pragma unroll
    for (int T = 0; T < 2; ++T) {
      const __bf16* krow = kvb
          + ((size_t)(bb * SEQ + s0 + T * 16 + lr)) * (2 * HSIZE) + h * ADIM;
      v16bf kf0 = make_frag(krow + kbA,      krow + 16 + kbA);      // d in [0,32)
      v16bf kf1 = make_frag(krow + 32 + kbA, krow + 48 + kbA);      // d in [32,64)
      v8f zero = {};
      v8f ct = __builtin_amdgcn_wmma_f32_16x16x32_bf16(
          false, qa[0], false, kf0, (short)0, zero, false, false);
      ct = __builtin_amdgcn_wmma_f32_16x16x32_bf16(
          false, qa[1], false, kf1, (short)0, ct, false, false);
#pragma unroll
      for (int r = 0; r < 8; ++r) {
        int m = r + mh;
        float sc = ct[r] * scale;
        if (Ms[(wid * 16 + m) * 32 + T * 16 + lr]) sc = NEG_INF;
        float u = sc + nb;
        u = u > 0.0f ? u : 0.0f;
        u = u * u;
        dsum[r] += u;
        Ts[wid][m * LDS_STRIDE + T * 16 + lr] = (__bf16)u;
      }
    }
    asm volatile("" ::: "memory");   // keep per-wave LDS t-store before t-load

    // t A-frag (16x32 over s_local) and o += t @ V
    v16bf ta = make_frag(&Ts[wid][lr * LDS_STRIDE + kbA],
                         &Ts[wid][lr * LDS_STRIDE + 16 + kbA]);
#pragma unroll
    for (int j = 0; j < 4; ++j) {
      v16bf vf = make_frag(&Vt[(j * 16 + lr) * LDS_STRIDE + kbA],
                           &Vt[(j * 16 + lr) * LDS_STRIDE + 16 + kbA]);
      oacc[j] = __builtin_amdgcn_wmma_f32_16x16x32_bf16(
          false, ta, false, vf, (short)0, oacc[j], false, false);
    }
  }

  // Denominator reduction across the 16 lanes sharing each row group
#pragma unroll
  for (int r = 0; r < 8; ++r) {
    float v = dsum[r];
    v += __shfl_xor(v, 1, 32);
    v += __shfl_xor(v, 2, 32);
    v += __shfl_xor(v, 4, 32);
    v += __shfl_xor(v, 8, 32);
    dsum[r] = 1.0f / (v + EPSF);
  }

#pragma unroll
  for (int j = 0; j < 4; ++j)
#pragma unroll
    for (int r = 0; r < 8; ++r) {
      int m = r + mh;
      ob[((size_t)(bb * NQ + q0 + m)) * HSIZE + h * ADIM + j * 16 + lr] =
          (__bf16)(oacc[j][r] * dsum[r]);
    }
}

// ---------------------------------------------------------------------------
// Launch
// ---------------------------------------------------------------------------
extern "C" void kernel_launch(void* const* d_in, const int* in_sizes, int n_in,
                              void* d_out, int out_size, void* d_ws, size_t ws_size,
                              hipStream_t stream) {
  const float* iQ  = (const float*)d_in[0];
  const float* iK  = (const float*)d_in[1];
  const unsigned char* mask = (const unsigned char*)d_in[2];
  const float* Wq  = (const float*)d_in[3];
  const float* Wkv = (const float*)d_in[4];
  const float* Wo  = (const float*)d_in[5];
  const float* nbias = (const float*)d_in[6];
  float* out = (float*)d_out;

  char* ws = (char*)d_ws;
  size_t off = 0;
  auto alloc = [&](size_t elems) -> __bf16* {
    __bf16* p = (__bf16*)(ws + off);
    off = (off + elems * sizeof(__bf16) + 255) & ~(size_t)255;
    return p;
  };

  const int NQI  = BATCH * NQ * ISIZE;      // 4M
  const int NKI  = BATCH * SEQ * ISIZE;     // 4M
  const int NWQ  = HSIZE * ISIZE;           // 1M
  const int NWKV = 2 * HSIZE * ISIZE;       // 2M
  const int NWO  = OSIZE * HSIZE;           // 1M

  __bf16* iQb  = alloc(NQI);
  __bf16* iKb  = alloc(NKI);
  __bf16* Wqb  = alloc(NWQ);
  __bf16* Wkvb = alloc(NWKV);
  __bf16* Wob  = alloc(NWO);
  __bf16* qb   = alloc((size_t)BATCH * NQ * HSIZE);       // (B*NQ, HSIZE)
  __bf16* kvb  = alloc((size_t)BATCH * SEQ * 2 * HSIZE);  // (B*SEQ, 2*HSIZE)
  __bf16* obuf = alloc((size_t)BATCH * NQ * HSIZE);

  cvt_bf16<<<2048, 256, 0, stream>>>(iQ,  iQb,  NQI);
  cvt_bf16<<<2048, 256, 0, stream>>>(iK,  iKb,  NKI);
  cvt_bf16<<<1024, 256, 0, stream>>>(Wq,  Wqb,  NWQ);
  cvt_bf16<<<1024, 256, 0, stream>>>(Wkv, Wkvb, NWKV);
  cvt_bf16<<<1024, 256, 0, stream>>>(Wo,  Wob,  NWO);

  dim3 blk(256);
  // q = iQ @ Wq^T : (4096x1024) x (1024x1024)^T -> bf16
  gemm_nt<false><<<dim3(32, 8), blk, 0, stream>>>(
      iQb, Wqb, nullptr, qb, BATCH * NQ, HSIZE, ISIZE);
  // kv = iK @ Wkv^T : (4096x1024) x (2048x1024)^T -> bf16
  gemm_nt<false><<<dim3(32, 16), blk, 0, stream>>>(
      iKb, Wkvb, nullptr, kvb, BATCH * SEQ, 2 * HSIZE, ISIZE);
  // fused attention -> obuf bf16
  attn_sqrelu<<<dim3(BATCH * NUM_HEAD, NQ / 64), dim3(128), 0, stream>>>(
      qb, kvb, mask, nbias, obuf);
  // out = o @ Wo^T -> f32
  gemm_nt<true><<<dim3(32, 8), blk, 0, stream>>>(
      obuf, Wob, out, nullptr, BATCH * NQ, OSIZE, HSIZE);
}
